// Linear4bit_18021682774288
// MI455X (gfx1250) — matrix operand order repaired and checked by reading, not compiled, not run on hardware
//
#include <hip/hip_runtime.h>

// Problem constants (from reference)
#define IN_F    4096
#define OUT_F   11008
#define GS      128
#define NG      32            // IN_F / GS
#define M_TOTAL 8192          // BATCH * SEQ

// Tiling
#define BM 128
#define BN 128
#define LDA 136               // 128 + 8 halves pad -> conflict-free ds_load_b128
#define LDB 136

typedef __attribute__((ext_vector_type(16))) _Float16 v16h;
typedef __attribute__((ext_vector_type(8)))  float    v8f;
typedef __attribute__((ext_vector_type(8)))  _Float16 h8;
typedef __attribute__((ext_vector_type(4)))  _Float16 h4;
typedef __attribute__((ext_vector_type(4)))  float    f4;
typedef __attribute__((ext_vector_type(4)))  int      i4;

__global__ __launch_bounds__(256)
void linear4bit_wmma(const float* __restrict__ x,
                     const int*   __restrict__ wp,
                     const float* __restrict__ scales,
                     const float* __restrict__ zeros,
                     const float* __restrict__ bias,
                     float* __restrict__ out)
{
    __shared__ _Float16 As[BM * LDA];   // x tile, f16, [m][k]
    __shared__ _Float16 Bs[BN * LDB];   // dequantized W tile, f16, [n][k]

    const int tid  = threadIdx.x;
    const int lane = tid & 31;
    const int wave = tid >> 5;          // 0..7
    const int wm   = wave >> 2;         // 0..1  (64-row sub-tile)
    const int wn   = wave & 3;          // 0..3  (32-col sub-tile)
    const int bn0  = blockIdx.x * BN;   // output-feature base
    const int bm0  = blockIdx.y * BM;   // token base

    v8f acc[4][2] = {};

    // A staging: 256 threads x 4 floats = 8 rows of 128 per iteration
    const int a_row = tid >> 5;            // 0..7
    const int a_col = (tid & 31) * 4;      // 0..124
    // B staging: 2 threads per output row, each handles 64 k (32 packed ints)
    const int  b_row  = tid >> 1;          // 0..127
    const int  b_kh   = (tid & 1) * 64;    // k-half within group
    const long o_glob = (long)bn0 + b_row;

    const float* x_base = x  + (long)bm0 * IN_F;
    const int*   w_base = wp + o_glob * (IN_F / 2) + (b_kh >> 1);

    const int l15 = lane & 15;
    const int lh  = lane >> 4;

    for (int g = 0; g < NG; ++g) {
        __syncthreads();   // previous tile fully consumed

        // ---- stage A: fp32 -> f16 into LDS ----
        {
            const float* xg = x_base + g * GS + a_col;
            #pragma unroll
            for (int it = 0; it < 16; ++it) {
                const int m = a_row + it * 8;
                f4 v = *(const f4*)(xg + (long)m * IN_F);
                h4 h;
                h.x = (_Float16)v.x; h.y = (_Float16)v.y;
                h.z = (_Float16)v.z; h.w = (_Float16)v.w;
                *(h4*)(&As[m * LDA + a_col]) = h;
            }
        }

        // ---- stage B: unpack nibbles + per-group affine dequant -> f16 ----
        {
            const float s   = scales[o_glob * NG + g];
            const float nzs = -zeros[o_glob * NG + g] * s;   // w = nib*s + nzs
            const int* wrow = w_base + g * (GS / 2);
            _Float16*  brow = &Bs[b_row * LDB + b_kh];
            #pragma unroll
            for (int it = 0; it < 8; ++it) {
                i4 p = *(const i4*)(wrow + it * 4);
                h8 hv;
                // reference stacks [high, low]: k=2j -> high nibble, k=2j+1 -> low
                hv[0] = (_Float16)fmaf((float)((p.x >> 4) & 15), s, nzs);
                hv[1] = (_Float16)fmaf((float)( p.x       & 15), s, nzs);
                hv[2] = (_Float16)fmaf((float)((p.y >> 4) & 15), s, nzs);
                hv[3] = (_Float16)fmaf((float)( p.y       & 15), s, nzs);
                hv[4] = (_Float16)fmaf((float)((p.z >> 4) & 15), s, nzs);
                hv[5] = (_Float16)fmaf((float)( p.z       & 15), s, nzs);
                hv[6] = (_Float16)fmaf((float)((p.w >> 4) & 15), s, nzs);
                hv[7] = (_Float16)fmaf((float)( p.w       & 15), s, nzs);
                *(h8*)(brow + it * 8) = hv;
            }
            if (g + 1 < NG) {   // warm L2/WGP$ for next group (global_prefetch_b8)
                __builtin_prefetch(wrow + GS / 2, 0, 0);
                __builtin_prefetch(x_base + (long)a_row * IN_F + (g + 1) * GS + a_col, 0, 0);
            }
        }

        __syncthreads();

        // ---- compute: 4 k-steps of 32, 8 WMMA each ----
        #pragma unroll
        for (int kt = 0; kt < 4; ++kt) {
            const int k0 = kt * 32;
            v16h a[4], b[2];
            // A frag (16-bit A 16x32 layout): lanes 0-15 K {0..7,16..23},
            // lanes 16-31 K {8..15,24..31}
            #pragma unroll
            for (int mi = 0; mi < 4; ++mi) {
                const _Float16* ap = &As[(wm * 64 + mi * 16 + l15) * LDA + k0 + lh * 8];
                h8 lo = *(const h8*)ap;
                h8 hi = *(const h8*)(ap + 16);
                a[mi] = __builtin_shufflevector(lo, hi,
                        0,1,2,3,4,5,6,7,8,9,10,11,12,13,14,15);
            }
            // B frag (32x16 K x N): lanes 0-15 K {0..7,16..23},
            // lanes 16-31 K {8..15,24..31} (matches A), Bs is [n][k]
            #pragma unroll
            for (int ni = 0; ni < 2; ++ni) {
                const _Float16* bp = &Bs[(wn * 32 + ni * 16 + l15) * LDB + k0 + lh * 8];
                h8 lo = *(const h8*)bp;
                h8 hi = *(const h8*)(bp + 16);
                b[ni] = __builtin_shufflevector(lo, hi,
                        0,1,2,3,4,5,6,7,8,9,10,11,12,13,14,15);
            }
            #pragma unroll
            for (int mi = 0; mi < 4; ++mi)
                #pragma unroll
                for (int ni = 0; ni < 2; ++ni)
                    acc[mi][ni] = __builtin_amdgcn_wmma_f32_16x16x32_f16(
                        false, a[mi], false, b[ni],
                        (short)0, acc[mi][ni], false, false);
        }
    }

    // ---- epilogue: C/D layout VGPR r -> M=r (lanes 0-15) / M=8+r (lanes 16-31) ----
    #pragma unroll
    for (int ni = 0; ni < 2; ++ni) {
        const int   n  = bn0 + wn * 32 + ni * 16 + l15;
        const float bv = bias[n];
        #pragma unroll
        for (int mi = 0; mi < 4; ++mi) {
            const long m0 = (long)bm0 + wm * 64 + mi * 16 + lh * 8;
            float* op = out + m0 * OUT_F + n;
            #pragma unroll
            for (int r = 0; r < 8; ++r)
                op[(long)r * OUT_F] = acc[mi][ni][r] + bv;
        }
    }
}

extern "C" void kernel_launch(void* const* d_in, const int* in_sizes, int n_in,
                              void* d_out, int out_size, void* d_ws, size_t ws_size,
                              hipStream_t stream) {
    const float* x      = (const float*)d_in[0];
    const int*   wp     = (const int*)  d_in[1];
    const float* scales = (const float*)d_in[2];
    const float* zeros  = (const float*)d_in[3];
    const float* bias   = (const float*)d_in[4];
    float*       out    = (float*)d_out;

    dim3 grid(OUT_F / BN, M_TOTAL / BM);   // 86 x 64 tiles
    linear4bit_wmma<<<grid, 256, 0, stream>>>(x, wp, scales, zeros, bias, out);
}
